// WindowAttention_53223234732397
// MI455X (gfx1250) — compile-verified, matrix operand order checked by reading, not run on hardware
//
#include <hip/hip_runtime.h>
#include <hip/hip_bf16.h>

typedef __attribute__((ext_vector_type(16))) _Float16 v16h;
typedef __attribute__((ext_vector_type(8)))  _Float16 v8h;
typedef __attribute__((ext_vector_type(4)))  _Float16 v4h;
typedef __attribute__((ext_vector_type(8)))  float    v8f;
typedef __attribute__((ext_vector_type(4)))  unsigned v4u;
typedef __attribute__((ext_vector_type(8)))  int      v8i;
typedef __attribute__((ext_vector_type(4)))  int      v4i;

#define N_TOK   64
#define DIM     256
#define HEADS   8
#define HD      32
#define SCALE_Q 0.17677669529663689f   // 1/sqrt(32)

// ---------------- LDS layout (bytes) ----------------
// mask  f32 [64][68]          :      0 .. 17408   (filled by TDM tensor_load_to_lds w/ HW padding)
// x/O   f16 [64][264]         :  17408 .. 51200   (x for QKV; reused for head outputs O)
// per-wave (8 waves, 14848 B each) at 51200:
//   q   f16 [64][40]  (+0)      -- overlaid later by P f16 [64][72]
//   k   f16 [64][40]  (+5120)
//   vT  f16 [32][72]  (+10240)
// out   f32 [64][264]         :  51200 .. 118784  (overlays all per-wave regions after 2nd barrier)
#define MASK_OFF    0
#define MSTR        68
#define X_OFF       17408
#define XS          264
#define WAVE_OFF    51200
#define WAVE_BYTES  14848
#define QS          40
#define VS          72
#define PS          72
#define OSTR        264
#define SMEM_BYTES  169984

// ---------------- ws layout ----------------
// wq_h  : _Float16[65536]             @ half-offset 0
// wkv_h : _Float16[131072]            @ half-offset 65536
// wp_h  : _Float16[65536]             @ half-offset 196608
// biasH : float[8*64*64]              @ byte 524288  (bias_table[rel_idx] pre-gathered per head)

__device__ __forceinline__ v8f wmma16(v16h a, v16h b, v8f c) {
  return __builtin_amdgcn_wmma_f32_16x16x32_f16(false, a, false, b, (short)0, c, false, false);
}

// Load one 16(row)x32(col) f16 operand tile from a row-major buffer (stride in halves,
// stride must be a multiple of 8). Produces the CDNA5 16-bit A-operand layout:
//   lane L: row = L&15 ; VGPR v: K = (v>>2)*16 + (L>>4)*8 + (v&3)*2 + {0,1}
// -> two aligned 16-byte loads per lane.
__device__ __forceinline__ v16h loadFragRM(const _Float16* base, int stride) {
  const int lane = threadIdx.x & 31;
  const _Float16* p = base + (lane & 15) * stride + ((lane >> 4) << 3);
  v8h lo = *(const v8h*)(p);
  v8h hi = *(const v8h*)(p + 16);
  v16h r;
#pragma unroll
  for (int i = 0; i < 8; ++i) { r[i] = lo[i]; r[i + 8] = hi[i]; }
  return r;
}

// -------- prep: f32->f16 weights + pre-gather relative-position bias per head --------
__global__ void wattn_prep_kernel(const float* __restrict__ wq,
                                  const float* __restrict__ wkv,
                                  const float* __restrict__ wp,
                                  const float* __restrict__ bias_table,
                                  const int*   __restrict__ rel_idx,
                                  _Float16* __restrict__ wsH,
                                  float* __restrict__ biasH) {
  int idx = blockIdx.x * blockDim.x + threadIdx.x;
  if (idx < 65536)        wsH[idx] = (_Float16)wq[idx];
  else if (idx < 196608)  wsH[idx] = (_Float16)wkv[idx - 65536];
  else if (idx < 262144)  wsH[idx] = (_Float16)wp[idx - 196608];
  if (idx < HEADS * N_TOK * N_TOK) {
    int h = idx >> 12, mn = idx & 4095;             // [h][m*64+n]
    biasH[idx] = bias_table[rel_idx[mn] * HEADS + h];
  }
}

// -------- main: one block per window, one wave per head --------
__global__ void __launch_bounds__(256, 1)
wattn_main_kernel(const float* __restrict__ x,
                  const float* __restrict__ mask,
                  const float* __restrict__ bq,
                  const float* __restrict__ bkv,
                  const float* __restrict__ bp,
                  const _Float16* __restrict__ wsH,
                  const float* __restrict__ biasH,
                  float* __restrict__ out) {
  extern __shared__ char smem[];
  const int tid  = threadIdx.x;
  const int blk  = blockIdx.x;
  const int wave = tid >> 5;
  const int h    = wave;
  const int lane = tid & 31;
  const int nl   = lane & 15;         // n within 16x16 tile
  const int mo   = (lane >> 4) * 8;   // m offset within tile (C/D layout)

  float*     mlds  = (float*)(smem + MASK_OFF);
  _Float16*  xlds  = (_Float16*)(smem + X_OFF);
  _Float16*  qlds  = (_Float16*)(smem + WAVE_OFF + wave * WAVE_BYTES);
  _Float16*  klds  = qlds + 5120 / 2;
  _Float16*  vtlds = qlds + 10240 / 2;
  _Float16*  plds  = qlds;            // overlay q+k after they are consumed
  _Float16*  olds  = xlds;            // overlay x after QKV stage
  float*     outlds = (float*)(smem + WAVE_OFF);   // overlays wave scratch after 2nd barrier

  const _Float16* wqh  = wsH;
  const _Float16* wkvh = wsH + 65536;
  const _Float16* wph  = wsH + 196608;

  // ---- TDM: DMA the 64x64 f32 mask tile into LDS with HW row padding (stride 68 dwords) ----
  if (wave == 0) {
    const float* mw = mask + (size_t)(blk & 4095) * (N_TOK * N_TOK);
    unsigned long long ga = (unsigned long long)(const void*)mw;
    unsigned ldsA = (unsigned)(size_t)(smem + MASK_OFF);   // low 32 bits = LDS byte address
    // D# group0: count=1 | lds_addr | global_addr[56:0] | type=2
    v4u g0 = { 1u, ldsA, (unsigned)ga,
               ((unsigned)(ga >> 32) & 0x1FFFFFFu) | (2u << 30) };
    // D# group1: data_size=4B, pad_enable, pad_interval=64dw(code 5), pad_amount=4dw(code 3),
    //            tensor_dim0=64, tensor_dim1=64, tile_dim0=64, tile_dim1=64, dim0_stride=64
    v8i g1 = { (2 << 16) | (1 << 20) | (5 << 22) | (3 << 25),
               64 << 16,   // tensor_dim0[15:0] (bits 63:48)
               64 << 16,   // tensor_dim1[15:0] (bits 95:80); tensor_dim0[31:16]=0
               64 << 16,   // tile_dim0 (bits 127:112); tensor_dim1[31:16]=0
               64,         // tile_dim1 (bits 143:128); tile_dim2=0
               64,         // tensor_dim0_stride[31:0]
               0, 0 };
    v4i gz = { 0, 0, 0, 0 };
#if defined(__clang_major__) && (__clang_major__ >= 23)
    v8i gz8 = { 0, 0, 0, 0, 0, 0, 0, 0 };
    __builtin_amdgcn_tensor_load_to_lds(g0, g1, gz, gz, gz8, 0);
#else
    __builtin_amdgcn_tensor_load_to_lds(g0, g1, gz, gz, 0);
#endif
  }

  // ---- cooperative stage: x tile (f32 -> f16) into LDS ----
  {
    const float* xb = x + (size_t)blk * (N_TOK * DIM);
#pragma unroll
    for (int it = 0; it < 16; ++it) {
      int i = (tid + it * 256) * 4;                 // 16384 floats
      float4 f = *(const float4*)(xb + i);
      v4h hv = { (_Float16)f.x, (_Float16)f.y, (_Float16)f.z, (_Float16)f.w };
      *(v4h*)(xlds + (i >> 8) * XS + (i & 255)) = hv;
    }
  }
  if (wave == 0) __builtin_amdgcn_s_wait_tensorcnt(0);
  __syncthreads();

  // ---- QKV projections for this head: [64,256] x [256,32] per matrix ----
#pragma unroll 1
  for (int mt = 0; mt < 4; ++mt) {
    v16h afr[8];
#pragma unroll
    for (int kt = 0; kt < 8; ++kt)
      afr[kt] = loadFragRM(xlds + mt * 16 * XS + kt * 32, XS);
#pragma unroll
    for (int o = 0; o < 6; ++o) {                   // q0,q1,k0,k1,v0,v1
      const int which = o >> 1, nt = o & 1;
      const _Float16* wrow;
      const float*    brow;
      if (which == 0)      { wrow = wqh  + (h * 32 + nt * 16) * DIM;        brow = bq  + h * 32 + nt * 16; }
      else if (which == 1) { wrow = wkvh + (h * 32 + nt * 16) * DIM;        brow = bkv + h * 32 + nt * 16; }
      else                 { wrow = wkvh + (DIM + h * 32 + nt * 16) * DIM;  brow = bkv + DIM + h * 32 + nt * 16; }
      v8f acc = {};
#pragma unroll
      for (int kt = 0; kt < 8; ++kt)
        acc = wmma16(afr[kt], loadFragRM(wrow + kt * 32, DIM), acc);
      const float bv = brow[nl];
      if (which == 0) {
#pragma unroll
        for (int r = 0; r < 8; ++r)
          qlds[(mt * 16 + mo + r) * QS + nt * 16 + nl] = (_Float16)((acc[r] + bv) * SCALE_Q);
      } else if (which == 1) {
#pragma unroll
        for (int r = 0; r < 8; ++r)
          klds[(mt * 16 + mo + r) * QS + nt * 16 + nl] = (_Float16)(acc[r] + bv);
      } else {                                       // store V transposed: vT[d][n]
#pragma unroll
        for (int r = 0; r < 8; ++r)
          vtlds[(nt * 16 + nl) * VS + mt * 16 + mo + r] = (_Float16)(acc[r] + bv);
      }
    }
  }

  // ---- S = (q*scale) @ k^T  (K = 32 = one WMMA depth) ----
  v16h qfr[4], kfr[4];
#pragma unroll
  for (int t = 0; t < 4; ++t) qfr[t] = loadFragRM(qlds + t * 16 * QS, QS);
#pragma unroll
  for (int t = 0; t < 4; ++t) kfr[t] = loadFragRM(klds + t * 16 * QS, QS);

  v8f s[4][4];
#pragma unroll
  for (int mt = 0; mt < 4; ++mt)
#pragma unroll
    for (int nt = 0; nt < 4; ++nt) {
      v8f z = {};
      s[mt][nt] = wmma16(qfr[mt], kfr[nt], z);
    }

  // ---- add relative-position bias (per head) + window mask ----
  {
    const float* bh = biasH + h * (N_TOK * N_TOK);
#pragma unroll
    for (int mt = 0; mt < 4; ++mt)
#pragma unroll
      for (int nt = 0; nt < 4; ++nt)
#pragma unroll
        for (int r = 0; r < 8; ++r) {
          const int m = mt * 16 + mo + r, n = nt * 16 + nl;
          s[mt][nt][r] += bh[m * N_TOK + n] + mlds[m * MSTR + n];
        }
  }

  // ---- softmax over rows; rows live across 16-lane halves in the C layout ----
#pragma unroll 1
  for (int mt = 0; mt < 4; ++mt) {
    float rmax[8], rinv[8];
#pragma unroll
    for (int r = 0; r < 8; ++r) {
      float v = fmaxf(fmaxf(s[mt][0][r], s[mt][1][r]), fmaxf(s[mt][2][r], s[mt][3][r]));
#pragma unroll
      for (int xm = 1; xm < 16; xm <<= 1) v = fmaxf(v, __shfl_xor(v, xm, 32));
      rmax[r] = v; rinv[r] = 0.0f;
    }
#pragma unroll
    for (int nt = 0; nt < 4; ++nt)
#pragma unroll
      for (int r = 0; r < 8; ++r) {
        float e = __expf(s[mt][nt][r] - rmax[r]);
        s[mt][nt][r] = e; rinv[r] += e;
      }
#pragma unroll
    for (int r = 0; r < 8; ++r) {
      float v = rinv[r];
#pragma unroll
      for (int xm = 1; xm < 16; xm <<= 1) v += __shfl_xor(v, xm, 32);
      rinv[r] = 1.0f / v;
    }
#pragma unroll
    for (int nt = 0; nt < 4; ++nt)
#pragma unroll
      for (int r = 0; r < 8; ++r)
        plds[(mt * 16 + mo + r) * PS + nt * 16 + nl] = (_Float16)(s[mt][nt][r] * rinv[r]);
  }

  // ---- O = P @ V  (64x64 @ 64x32) ----
  v16h vb[2][2];
#pragma unroll
  for (int dt = 0; dt < 2; ++dt)
#pragma unroll
    for (int ks = 0; ks < 2; ++ks)
      vb[dt][ks] = loadFragRM(vtlds + dt * 16 * VS + ks * 32, VS);

  v8f oacc[4][2];
#pragma unroll
  for (int mt = 0; mt < 4; ++mt) {
    v16h a0 = loadFragRM(plds + mt * 16 * PS, PS);
    v16h a1 = loadFragRM(plds + mt * 16 * PS + 32, PS);
#pragma unroll
    for (int dt = 0; dt < 2; ++dt) {
      v8f acc = {};
      acc = wmma16(a0, vb[dt][0], acc);
      acc = wmma16(a1, vb[dt][1], acc);
      oacc[mt][dt] = acc;
    }
  }

  __syncthreads();   // everyone is done reading xlds -> safe to overlay with O
#pragma unroll
  for (int mt = 0; mt < 4; ++mt)
#pragma unroll
    for (int dt = 0; dt < 2; ++dt)
#pragma unroll
      for (int r = 0; r < 8; ++r)
        olds[(mt * 16 + mo + r) * XS + h * 32 + dt * 16 + nl] = (_Float16)oacc[mt][dt][r];
  __syncthreads();   // all head outputs visible; per-wave q/k/vT/P regions now dead

  // ---- out = O_all @ wp^T + bp ; wave h produces output columns [h*32, h*32+32) ----
  // Results staged f32 in LDS, then cooperatively stored with b128 for coalescing.
#pragma unroll 1
  for (int mt = 0; mt < 4; ++mt) {
    v16h afr[8];
#pragma unroll
    for (int kt = 0; kt < 8; ++kt)
      afr[kt] = loadFragRM(olds + mt * 16 * XS + kt * 32, XS);
#pragma unroll
    for (int nt = 0; nt < 2; ++nt) {
      const _Float16* wrow = wph + (h * 32 + nt * 16) * DIM;
      v8f acc = {};
#pragma unroll
      for (int kt = 0; kt < 8; ++kt)
        acc = wmma16(afr[kt], loadFragRM(wrow + kt * 32, DIM), acc);
      const float bv = bp[h * 32 + nt * 16 + nl];
#pragma unroll
      for (int r = 0; r < 8; ++r)
        outlds[(mt * 16 + mo + r) * OSTR + h * 32 + nt * 16 + nl] = acc[r] + bv;
    }
  }
  __syncthreads();

  // ---- fully-coalesced b128 stores of the window's 64x256 f32 output ----
  {
    float* outb = out + (size_t)blk * (N_TOK * DIM);
#pragma unroll
    for (int it = 0; it < 16; ++it) {
      int i = (tid + it * 256) * 4;                 // 16384 floats
      float4 f = *(const float4*)(outlds + (i >> 8) * OSTR + (i & 255));
      *(float4*)(outb + i) = f;
    }
  }
}

extern "C" void kernel_launch(void* const* d_in, const int* in_sizes, int n_in,
                              void* d_out, int out_size, void* d_ws, size_t ws_size,
                              hipStream_t stream) {
  const float* x    = (const float*)d_in[0];
  const float* mask = (const float*)d_in[1];
  const float* wq   = (const float*)d_in[2];
  const float* bq   = (const float*)d_in[3];
  const float* wkv  = (const float*)d_in[4];
  const float* bkv  = (const float*)d_in[5];
  const float* wp   = (const float*)d_in[6];
  const float* bp   = (const float*)d_in[7];
  const float* bt   = (const float*)d_in[8];
  const int*   ri   = (const int*)d_in[9];

  _Float16* wsH   = (_Float16*)d_ws;
  float*    biasH = (float*)((char*)d_ws + 524288);
  float*    outp  = (float*)d_out;

  wattn_prep_kernel<<<1024, 256, 0, stream>>>(wq, wkv, wp, bt, ri, wsH, biasH);
  wattn_main_kernel<<<8192, 256, SMEM_BYTES, stream>>>(x, mask, bq, bkv, bp, wsH, biasH, outp);
  (void)in_sizes; (void)n_in; (void)out_size; (void)ws_size;
}
